// WindowedMSA_14783277433183
// MI455X (gfx1250) — matrix operand order repaired and checked by reading, not compiled, not run on hardware
//
#include <hip/hip_runtime.h>
#include <hip/hip_bf16.h>

typedef __attribute__((ext_vector_type(16))) _Float16 v16h;
typedef __attribute__((ext_vector_type(8)))  float    v8f;

#define DIM   192
#define NH    6
#define HD    32
#define WSZ   8
#define TT    64          // WSZ*WSZ tokens per window
#define NWIN  4096        // 4 * 32 * 32 windows
#define NHW   (NWIN * NH) // 24576 head-windows
#define ROWS  262144      // B*H*W tokens
#define VTS   72          // padded stride (halves) for transposed V rows

// ---------------------------------------------------------------------------
// Kernel A: QKV projection  (x[262144,192] @ qkv_w^T[192,576] + b)
// f16 WMMA, f32 accumulate; epilogue scatters q/k/v into
// (window, head, t, d) f16 layout for the attention kernel.
// grid (16384, 9), block 128 (4 waves); one 16x16 tile per wave.
// ---------------------------------------------------------------------------
__global__ __launch_bounds__(128)
void qkv_wmma_kernel(const float* __restrict__ x, const float* __restrict__ w,
                     const float* __restrict__ bias, _Float16* __restrict__ qB,
                     size_t QS)
{
    const int lane = threadIdx.x & 31;
    const int wave = threadIdx.x >> 5;
    const int m0   = blockIdx.x << 4;
    const int n0   = ((blockIdx.y << 2) + wave) << 4;   // 36 N-tiles of 16
    const int mrow = lane & 15;
    const int hi   = lane >> 4;                         // half-wave select
    const int akb  = hi << 3;                           // A K sub-block 0/8

    const float* xr = x + (size_t)(m0 + mrow) * DIM;
    const float* wr = w + (size_t)(n0 + mrow) * DIM;

    v8f c = {};
#pragma unroll
    for (int k0 = 0; k0 < DIM; k0 += 32) {
        v16h a, b;
#pragma unroll
        for (int j = 0; j < 8; ++j) {
            a[j]     = (_Float16)xr[k0 + akb + j];       // K = k0+akb+j
            a[8 + j] = (_Float16)xr[k0 + 16 + akb + j];  // K = k0+16+akb+j
        }
#pragma unroll
        for (int j = 0; j < 8; ++j) {
            b[j]     = (_Float16)wr[k0 + akb + j];       // K = k0+akb+j
            b[8 + j] = (_Float16)wr[k0 + 16 + akb + j];  // K = k0+16+akb+j
        }
        c = __builtin_amdgcn_wmma_f32_16x16x32_f16(false, a, false, b,
                                                   (short)0, c, false, false);
    }

    // epilogue: scatter into window/head layout
    const int col  = n0 + mrow;
    const int mat  = col / DIM;        // 0=q 1=k 2=v (uniform per wave)
    const int ch   = col % DIM;
    const int head = ch >> 5;
    const int d    = ch & 31;
    const float bv = bias[col];
    _Float16* dst  = qB + (size_t)mat * QS;
    const int mb   = hi << 3;
#pragma unroll
    for (int v = 0; v < 8; ++v) {
        int r   = m0 + mb + v;                 // global token row
        int b_  = r >> 16;                     // batch (H*W = 65536)
        int p   = r & 65535;
        int y   = p >> 8, xx = p & 255;
        int win = ((b_ << 5) + (y >> 3)) * 32 + (xx >> 3);
        int t   = ((y & 7) << 3) | (xx & 7);
        size_t idx = (((size_t)(win * NH + head) << 6) + t) * HD + d;
        dst[idx] = (_Float16)(c[v] + bv);
    }
}

// ---------------------------------------------------------------------------
// Kernel B: fused windowed attention + LePE.
// One block (4 waves) per head-window. Q/K staged row-major in LDS; V staged
// TRANSPOSED (VsT[d][t], stride 72 halves: bank-conflict-free 16-lane column
// access, 16B-aligned rows) so PV B-fragments are contiguous ds_load_b128s.
// S = QK^T via WMMA (HD=32 => single K-step), in-register softmax with
// 16-lane shfl_xor row reductions, normalized P staged f16 in LDS,
// O = P V via WMMA, LePE depthwise 3x3 from VsT fused into the epilogue,
// scatter to (B, N, C) f16 for the output projection.
// ---------------------------------------------------------------------------
__global__ __launch_bounds__(128)
void attn_wmma_kernel(const _Float16* __restrict__ qB, const _Float16* __restrict__ kB,
                      const _Float16* __restrict__ vB,
                      const float* __restrict__ pw, const float* __restrict__ pb,
                      _Float16* __restrict__ oB)
{
    __shared__ __align__(16) _Float16 Qs[TT * HD];
    __shared__ __align__(16) _Float16 Ks[TT * HD];
    __shared__ __align__(16) _Float16 VsT[HD * VTS];   // V transposed: [d][t]
    __shared__ __align__(16) _Float16 Ps[TT * TT];

    const int tid  = threadIdx.x;
    const int lane = tid & 31;
    const int wave = tid >> 5;
    const int hw   = blockIdx.x;         // head-window = win*NH + head
    const int head = hw % NH;
    const int win  = hw / NH;

    // cooperative loads: Q, K straight (128-bit), V loaded 128-bit and
    // transposed into VsT via 2B LDS stores (paid once per block)
    {
        const uint4* gq = (const uint4*)(qB + (size_t)hw * (TT * HD));
        const uint4* gk = (const uint4*)(kB + (size_t)hw * (TT * HD));
        const uint4* gv = (const uint4*)(vB + (size_t)hw * (TT * HD));
        uint4* sq = (uint4*)Qs; uint4* sk = (uint4*)Ks;
        sq[tid] = gq[tid];  sq[tid + 128] = gq[tid + 128];
        sk[tid] = gk[tid];  sk[tid + 128] = gk[tid + 128];
#pragma unroll
        for (int half = 0; half < 2; ++half) {
            int i = tid + half * 128;            // uint4 index into V
            uint4 raw = gv[i];
            const _Float16* h = (const _Float16*)&raw;   // 8 halves along d
            int t  = i >> 2;                     // token row
            int d0 = (i & 3) << 3;               // starting channel
#pragma unroll
            for (int j = 0; j < 8; ++j)
                VsT[(d0 + j) * VTS + t] = h[j];
        }
    }
    __syncthreads();

    const int mrow = lane & 15;
    const int hi   = lane >> 4;
    const int akb  = hi << 3;
    const int r0   = wave << 4;          // this wave's 16 query rows

    // ---- S = Q K^T : 4 tiles of 16x16, single K=32 step each ----
    v16h aq;
    {
        const _Float16* q = Qs + (r0 + mrow) * HD;
#pragma unroll
        for (int j = 0; j < 8; ++j) { aq[j] = q[akb + j]; aq[8 + j] = q[16 + akb + j]; }
    }
    v8f s[4];
#pragma unroll
    for (int nt = 0; nt < 4; ++nt) {
        v16h bk;
        const _Float16* kk = Ks + ((nt << 4) + mrow) * HD;
#pragma unroll
        for (int j = 0; j < 8; ++j) { bk[j] = kk[akb + j]; bk[8 + j] = kk[16 + akb + j]; }
        v8f z = {};
        s[nt] = __builtin_amdgcn_wmma_f32_16x16x32_f16(false, aq, false, bk,
                                                       (short)0, z, false, false);
    }

    // ---- row softmax (each C-tile row lives in one VGPR across a 16-lane half) ----
    const float scale = 0.17677669529663687f;  // 1/sqrt(HD)
#pragma unroll
    for (int v = 0; v < 8; ++v) {
        float x0 = s[0][v] * scale, x1 = s[1][v] * scale;
        float x2 = s[2][v] * scale, x3 = s[3][v] * scale;
        float m = fmaxf(fmaxf(x0, x1), fmaxf(x2, x3));
#pragma unroll
        for (int off = 8; off > 0; off >>= 1) m = fmaxf(m, __shfl_xor(m, off, 32));
        float e0 = __expf(x0 - m), e1 = __expf(x1 - m);
        float e2 = __expf(x2 - m), e3 = __expf(x3 - m);
        float sum = e0 + e1 + e2 + e3;
#pragma unroll
        for (int off = 8; off > 0; off >>= 1) sum += __shfl_xor(sum, off, 32);
        float rs = 1.0f / sum;
        _Float16* pr = Ps + (r0 + (hi << 3) + v) * TT + mrow;
        pr[0]  = (_Float16)(e0 * rs);
        pr[16] = (_Float16)(e1 * rs);
        pr[32] = (_Float16)(e2 * rs);
        pr[48] = (_Float16)(e3 * rs);
    }
    __syncthreads();

    // ---- O = P V : 2 N-tiles, K=64 in two 32-steps ----
    // B-fragment from VsT: B[k][n] = V[k][n] = VsT[n][k]  (K contiguous!)
    v8f o[2] = {};
#pragma unroll
    for (int k0 = 0; k0 < TT; k0 += 32) {
        v16h ap;
        const _Float16* p = Ps + (r0 + mrow) * TT + k0;
#pragma unroll
        for (int j = 0; j < 8; ++j) { ap[j] = p[akb + j]; ap[8 + j] = p[16 + akb + j]; }
#pragma unroll
        for (int nt = 0; nt < 2; ++nt) {
            v16h bvv;
            const _Float16* vv = VsT + ((nt << 4) + mrow) * VTS + k0;
#pragma unroll
            for (int j = 0; j < 8; ++j) { bvv[j] = vv[akb + j]; bvv[8 + j] = vv[16 + akb + j]; }
            o[nt] = __builtin_amdgcn_wmma_f32_16x16x32_f16(false, ap, false, bvv,
                                                           (short)0, o[nt], false, false);
        }
    }

    // ---- fused LePE (depthwise 3x3 over VsT) + scatter to (B,N,C) f16 ----
    const int b_ = win >> 10;
    const int wy = (win >> 5) & 31;
    const int wx = win & 31;
#pragma unroll
    for (int nt = 0; nt < 2; ++nt) {
        const int d = (nt << 4) + mrow;
        const _Float16* vcol = VsT + d * VTS;        // V[:, d], unit stride in t
        const float* pwc = pw + (size_t)(head * HD + d) * 9;
        const float bvp  = pb[head * HD + d];
#pragma unroll
        for (int v = 0; v < 8; ++v) {
            const int t  = r0 + (hi << 3) + v;
            const int ty = t >> 3, tx = t & 7;
            float acc = o[nt][v] + bvp;
#pragma unroll
            for (int ky = -1; ky <= 1; ++ky) {
#pragma unroll
                for (int kx = -1; kx <= 1; ++kx) {
                    int yy = ty + ky, xx = tx + kx;
                    if (yy >= 0 && yy < 8 && xx >= 0 && xx < 8)
                        acc += (float)vcol[(yy << 3) | xx]
                             * pwc[(ky + 1) * 3 + (kx + 1)];
                }
            }
            int gy = (wy << 3) + ty, gx = (wx << 3) + tx;
            size_t row = ((size_t)b_ << 16) + ((size_t)gy << 8) + gx;
            oB[row * DIM + head * HD + d] = (_Float16)acc;
        }
    }
}

// ---------------------------------------------------------------------------
// Kernel C: output projection  (o[262144,192] @ out_w^T[192,192] + b) -> f32
// grid (16384, 3), block 128 (4 waves), A already f16 in memory.
// ---------------------------------------------------------------------------
__global__ __launch_bounds__(128)
void proj_wmma_kernel(const _Float16* __restrict__ oB, const float* __restrict__ w,
                      const float* __restrict__ bias, float* __restrict__ out)
{
    const int lane = threadIdx.x & 31;
    const int wave = threadIdx.x >> 5;
    const int m0   = blockIdx.x << 4;
    const int n0   = ((blockIdx.y << 2) + wave) << 4;   // 12 N-tiles
    const int mrow = lane & 15;
    const int hi   = lane >> 4;
    const int akb  = hi << 3;

    const _Float16* ar = oB + (size_t)(m0 + mrow) * DIM;
    const float*    wr = w  + (size_t)(n0 + mrow) * DIM;

    v8f c = {};
#pragma unroll
    for (int k0 = 0; k0 < DIM; k0 += 32) {
        v16h a, b;
#pragma unroll
        for (int j = 0; j < 8; ++j) { a[j] = ar[k0 + akb + j]; a[8 + j] = ar[k0 + 16 + akb + j]; }
#pragma unroll
        for (int j = 0; j < 8; ++j) { b[j] = (_Float16)wr[k0 + akb + j]; b[8 + j] = (_Float16)wr[k0 + 16 + akb + j]; }
        c = __builtin_amdgcn_wmma_f32_16x16x32_f16(false, a, false, b,
                                                   (short)0, c, false, false);
    }
    const float bv = bias[n0 + mrow];
#pragma unroll
    for (int v = 0; v < 8; ++v) {
        int r = m0 + (hi << 3) + v;
        out[(size_t)r * DIM + n0 + mrow] = c[v] + bv;
    }
}

// ---------------------------------------------------------------------------
extern "C" void kernel_launch(void* const* d_in, const int* in_sizes, int n_in,
                              void* d_out, int out_size, void* d_ws, size_t ws_size,
                              hipStream_t stream)
{
    const float* x     = (const float*)d_in[0];
    const float* qkv_w = (const float*)d_in[1];
    const float* qkv_b = (const float*)d_in[2];
    const float* pe_w  = (const float*)d_in[3];
    const float* pe_b  = (const float*)d_in[4];
    const float* out_w = (const float*)d_in[5];
    const float* out_b = (const float*)d_in[6];
    (void)in_sizes; (void)n_in; (void)out_size; (void)ws_size;

    // workspace: q | k | v (window layout, f16) | o (token layout, f16)
    const size_t QS = (size_t)NHW * TT * HD;   // 50,331,648 elems each
    _Float16* qB = (_Float16*)d_ws;
    _Float16* kB = qB + QS;
    _Float16* vB = qB + 2 * QS;
    _Float16* oB = qB + 3 * QS;

    dim3 gA(ROWS / 16, (3 * DIM) / 64);        // (16384, 9)
    qkv_wmma_kernel<<<gA, 128, 0, stream>>>(x, qkv_w, qkv_b, qB, QS);

    attn_wmma_kernel<<<NHW, 128, 0, stream>>>(qB, kB, vB, pe_w, pe_b, oB);

    dim3 gC(ROWS / 16, DIM / 64);              // (16384, 3)
    proj_wmma_kernel<<<gC, 128, 0, stream>>>(oB, out_w, out_b, (float*)d_out);
}
